// GRU_89103391523544
// MI455X (gfx1250) — compile-verified, matrix operand order checked by reading, not run on hardware
//
#include <hip/hip_runtime.h>
#include <hip/hip_bf16.h>

typedef __attribute__((ext_vector_type(2)))  float    v2f;
typedef __attribute__((ext_vector_type(8)))  float    v8f;
typedef __attribute__((ext_vector_type(8)))  _Float16 v8h;
typedef __attribute__((ext_vector_type(16))) _Float16 v16h;

#define SEQg   70
#define BATCHg 128
#define HIDg   512
#define EMBg   512
#define VOCABg 10000
#define DEC_NT 5          // 5 N-tiles (16 cols each) per wave: 10000 = 80*125

__device__ __forceinline__ float sigmoidf_(float x) { return 1.0f / (1.0f + __expf(-x)); }

// out[n*K + k] = in[k*N + n]   (K x N row-major -> N x K row-major)
__global__ void transpose_f32(const float* __restrict__ in, float* __restrict__ out, int K, int N) {
    int idx = blockIdx.x * blockDim.x + threadIdx.x;
    if (idx >= K * N) return;
    int k = idx % K;
    int n = idx / K;
    out[idx] = in[(size_t)k * N + n];
}

__global__ void transpose_cvt_f16(const float* __restrict__ in, _Float16* __restrict__ out, int K, int N) {
    int idx = blockIdx.x * blockDim.x + threadIdx.x;
    if (idx >= K * N) return;
    int k = idx % K;
    int n = idx / K;
    out[idx] = (_Float16)in[(size_t)k * N + n];
}

__global__ void cvt_f32_f16(const float* __restrict__ in, _Float16* __restrict__ out, int n) {
    int idx = blockIdx.x * blockDim.x + threadIdx.x;
    if (idx >= n) return;
    out[idx] = (_Float16)in[idx];
}

// One wave computes a 16x32 tile (2 N-tiles) via v_wmma_f32_16x16x4_f32.
// A: M x K row-major f32 (or rows gathered from emb table via tokens)
// Bt: N x K row-major f32 (pre-transposed weights) -> B columns contiguous
// MODE 0: C = A@B        MODE 1: C = A@B + bias[n]
// MODE 2: GRU epilogue:  C = (1-z)*h + z*tanh(gwh + A@B)   (ldc = N = HID)
template <int MODE>
__global__ void gemm_wmma_f32(const float* __restrict__ A, const int* __restrict__ tokens,
                              const float* __restrict__ Bt, float* __restrict__ C,
                              int N, int K,
                              const float* __restrict__ bias,
                              const float* __restrict__ zbuf,
                              const float* __restrict__ hprev,
                              const float* __restrict__ gwh) {
    const int lane  = threadIdx.x;          // 0..31
    const int hi    = lane >> 4;            // 0: K-low pair, 1: K-high pair
    const int l16   = lane & 15;
    const int nBase = blockIdx.x * 32;
    const int mBase = blockIdx.y * 16;

    const int  rowA = mBase + l16;
    const long arow = tokens ? (long)tokens[rowA] : (long)rowA;
    const float* ap  = A  + arow * (long)K + hi * 2;             // A[m][k], A[m][k+1]
    const float* bp0 = Bt + (long)(nBase + l16) * K + hi * 2;    // col n,    K pair
    const float* bp1 = bp0 + (long)16 * K;                       // col n+16, K pair

    v8f acc0 = {}, acc1 = {};
#pragma unroll 4
    for (int k = 0; k < K; k += 4) {
        // load everything for this k-step first, then both WMMAs
        v2f av = *(const v2f*)(ap + k);
        v2f b0 = *(const v2f*)(bp0 + k);
        v2f b1 = *(const v2f*)(bp1 + k);
        acc0 = __builtin_amdgcn_wmma_f32_16x16x4_f32(false, av, false, b0,
                                                     (short)0, acc0, false, false);
        acc1 = __builtin_amdgcn_wmma_f32_16x16x4_f32(false, av, false, b1,
                                                     (short)0, acc1, false, false);
    }

#pragma unroll
    for (int nt = 0; nt < 2; ++nt) {
        v8f acc = nt ? acc1 : acc0;
        int n = nBase + nt * 16 + l16;
#pragma unroll
        for (int j = 0; j < 8; ++j) {
            int m = mBase + j + hi * 8;   // C/D layout: VGPR j -> M=j / M=8+j
            float v = acc[j];
            if (MODE == 0) {
                C[(long)m * N + n] = v;
            } else if (MODE == 1) {
                C[(long)m * N + n] = v + bias[n];
            } else {
                float z  = zbuf[m * HIDg + n];
                float h  = hprev[m * HIDg + n];
                float ht = tanhf(gwh[m * (3 * HIDg) + n] + v);
                C[(long)m * N + n] = (1.0f - z) * h + z * ht;
            }
        }
    }
}

// r = sigmoid(Wr + Ur), z = sigmoid(Wz + Uz); store z and r*h
__global__ void gru_gates(const float* __restrict__ g, const float* __restrict__ u,
                          const float* __restrict__ hprev,
                          float* __restrict__ zbuf, float* __restrict__ rh) {
    int i = blockIdx.x * blockDim.x + threadIdx.x;
    if (i >= BATCHg * HIDg) return;
    int m = i >> 9;
    int n = i & (HIDg - 1);
    float r = sigmoidf_(g[m * 3 * HIDg + n]        + u[m * 2 * HIDg + n]);
    float z = sigmoidf_(g[m * 3 * HIDg + HIDg + n] + u[m * 2 * HIDg + HIDg + n]);
    zbuf[i] = z;
    rh[i]   = r * hprev[i];
}

// Decoder: one wave computes a 16 x (16*DEC_NT) strip via v_wmma_f32_16x16x32_f16.
// A: M x K row-major f16 (pre-converted hidden states) — loaded once per k-step.
// All DEC_NT B operands are materialized into distinct registers BEFORE any WMMA,
// so all 12 loads issue as one clause and WMMAs retire under partial loadcnt waits.
__global__ void dec_gemm_wmma_f16(const _Float16* __restrict__ Ah,
                                  const _Float16* __restrict__ Bt,
                                  const float* __restrict__ bias,
                                  float* __restrict__ C, int N, int K) {
    const int lane   = threadIdx.x;
    const int hi     = lane >> 4;
    const int l16    = lane & 15;
    const int nBase0 = blockIdx.x * (16 * DEC_NT);
    const int mBase  = blockIdx.y * 16;

    const _Float16* ap = Ah + (long)(mBase + l16) * K + hi * 8;
    const _Float16* bp[DEC_NT];
#pragma unroll
    for (int nt = 0; nt < DEC_NT; ++nt)
        bp[nt] = Bt + (long)(nBase0 + nt * 16 + l16) * K + hi * 8;

    v8f acc[DEC_NT];
#pragma unroll
    for (int nt = 0; nt < DEC_NT; ++nt) acc[nt] = (v8f){};

#pragma unroll 2
    for (int k = 0; k < K; k += 32) {
        // 16-bit layout: lanes0-15 hold K={0..7,16..23}, lanes16-31 K={8..15,24..31}
        // ---- phase 1: issue ALL loads for this k-step ----
        v8h a0 = *(const v8h*)(ap + k);
        v8h a1 = *(const v8h*)(ap + k + 16);
        v8h b0[DEC_NT], b1[DEC_NT];
#pragma unroll
        for (int nt = 0; nt < DEC_NT; ++nt) {
            b0[nt] = *(const v8h*)(bp[nt] + k);
            b1[nt] = *(const v8h*)(bp[nt] + k + 16);
        }
        // ---- phase 2: 5 independent WMMA chains ----
        v16h a;
#pragma unroll
        for (int j = 0; j < 8; ++j) { a[j] = a0[j]; a[8 + j] = a1[j]; }
#pragma unroll
        for (int nt = 0; nt < DEC_NT; ++nt) {
            v16h b;
#pragma unroll
            for (int j = 0; j < 8; ++j) { b[j] = b0[nt][j]; b[8 + j] = b1[nt][j]; }
            acc[nt] = __builtin_amdgcn_wmma_f32_16x16x32_f16(false, a, false, b,
                                                             (short)0, acc[nt], false, false);
        }
    }

#pragma unroll
    for (int nt = 0; nt < DEC_NT; ++nt) {
        int n = nBase0 + nt * 16 + l16;
#pragma unroll
        for (int j = 0; j < 8; ++j) {
            int m = mBase + j + hi * 8;
            C[(long)m * N + n] = acc[nt][j] + bias[n];
        }
    }
}

extern "C" void kernel_launch(void* const* d_in, const int* in_sizes, int n_in,
                              void* d_out, int out_size, void* d_ws, size_t ws_size,
                              hipStream_t stream) {
    (void)in_sizes; (void)n_in; (void)out_size; (void)ws_size;
    const int*   inputs = (const int*)  d_in[0];   // (70,128)
    const float* hidden = (const float*)d_in[1];   // (2,128,512)
    const float* emb    = (const float*)d_in[2];   // (10000,512)
    const float* W_x0   = (const float*)d_in[3];   // (512,1536)
    const float* U_h0   = (const float*)d_in[4];   // (512,1024)
    const float* U_ht0  = (const float*)d_in[5];   // (512,512)
    const float* b0     = (const float*)d_in[6];   // (1536)
    const float* W_x1   = (const float*)d_in[7];
    const float* U_h1   = (const float*)d_in[8];
    const float* U_ht1  = (const float*)d_in[9];
    const float* b1     = (const float*)d_in[10];
    const float* dec_W  = (const float*)d_in[11];  // (512,10000)
    const float* dec_b  = (const float*)d_in[12];  // (10000)

    char*  ws  = (char*)d_ws;
    size_t off = 0;
    auto alloc = [&](size_t bytes) -> void* {
        void* p = ws + off;
        off += (bytes + 255) & ~(size_t)255;
        return p;
    };
    float* Wx0t  = (float*)alloc((size_t)1536 * 512 * 4);
    float* Uh0t  = (float*)alloc((size_t)1024 * 512 * 4);
    float* Uht0t = (float*)alloc((size_t)512  * 512 * 4);
    float* Wx1t  = (float*)alloc((size_t)1536 * 512 * 4);
    float* Uh1t  = (float*)alloc((size_t)1024 * 512 * 4);
    float* Uht1t = (float*)alloc((size_t)512  * 512 * 4);
    float* Hseq  = (float*)alloc((size_t)SEQg * BATCHg * HIDg * 4);   // all h1 states (f32)
    float* h0a   = (float*)alloc((size_t)BATCHg * HIDg * 4);
    float* h0b   = (float*)alloc((size_t)BATCHg * HIDg * 4);
    float* gbuf  = (float*)alloc((size_t)BATCHg * 3 * HIDg * 4);
    float* ubuf  = (float*)alloc((size_t)BATCHg * 2 * HIDg * 4);
    float* zbuf  = (float*)alloc((size_t)BATCHg * HIDg * 4);
    float* rhbuf = (float*)alloc((size_t)BATCHg * HIDg * 4);
    _Float16* decWt = (_Float16*)alloc((size_t)VOCABg * HIDg * 2);
    _Float16* Hh    = (_Float16*)alloc((size_t)SEQg * BATCHg * HIDg * 2);

    const int tpb = 256;
    transpose_f32<<<dim3((512 * 1536 + tpb - 1) / tpb), tpb, 0, stream>>>(W_x0,  Wx0t,  512, 1536);
    transpose_f32<<<dim3((512 * 1024 + tpb - 1) / tpb), tpb, 0, stream>>>(U_h0,  Uh0t,  512, 1024);
    transpose_f32<<<dim3((512 * 512  + tpb - 1) / tpb), tpb, 0, stream>>>(U_ht0, Uht0t, 512, 512);
    transpose_f32<<<dim3((512 * 1536 + tpb - 1) / tpb), tpb, 0, stream>>>(W_x1,  Wx1t,  512, 1536);
    transpose_f32<<<dim3((512 * 1024 + tpb - 1) / tpb), tpb, 0, stream>>>(U_h1,  Uh1t,  512, 1024);
    transpose_f32<<<dim3((512 * 512  + tpb - 1) / tpb), tpb, 0, stream>>>(U_ht1, Uht1t, 512, 512);
    transpose_cvt_f16<<<dim3((512 * VOCABg + tpb - 1) / tpb), tpb, 0, stream>>>(dec_W, decWt, 512, VOCABg);

    const float* h0p = hidden;                       // layer0 initial h
    const float* h1p = hidden + BATCHg * HIDg;       // layer1 initial h
    dim3 wblk(32);
    for (int t = 0; t < SEQg; ++t) {
        float* h0n = (t & 1) ? h0b : h0a;
        float* h1n = Hseq + (size_t)t * BATCHg * HIDg;
        // ---- layer 0: x_t gathered from emb via tokens ----
        gemm_wmma_f32<1><<<dim3(1536 / 32, BATCHg / 16), wblk, 0, stream>>>(
            emb, inputs + t * BATCHg, Wx0t, gbuf, 1536, EMBg, b0, nullptr, nullptr, nullptr);
        gemm_wmma_f32<0><<<dim3(1024 / 32, BATCHg / 16), wblk, 0, stream>>>(
            h0p, nullptr, Uh0t, ubuf, 1024, HIDg, nullptr, nullptr, nullptr, nullptr);
        gru_gates<<<dim3((BATCHg * HIDg) / tpb), tpb, 0, stream>>>(gbuf, ubuf, h0p, zbuf, rhbuf);
        gemm_wmma_f32<2><<<dim3(HIDg / 32, BATCHg / 16), wblk, 0, stream>>>(
            rhbuf, nullptr, Uht0t, h0n, HIDg, HIDg, nullptr, zbuf, h0p, gbuf + 2 * HIDg);
        // ---- layer 1: x = h0n ----
        gemm_wmma_f32<1><<<dim3(1536 / 32, BATCHg / 16), wblk, 0, stream>>>(
            h0n, nullptr, Wx1t, gbuf, 1536, HIDg, b1, nullptr, nullptr, nullptr);
        gemm_wmma_f32<0><<<dim3(1024 / 32, BATCHg / 16), wblk, 0, stream>>>(
            h1p, nullptr, Uh1t, ubuf, 1024, HIDg, nullptr, nullptr, nullptr, nullptr);
        gru_gates<<<dim3((BATCHg * HIDg) / tpb), tpb, 0, stream>>>(gbuf, ubuf, h1p, zbuf, rhbuf);
        gemm_wmma_f32<2><<<dim3(HIDg / 32, BATCHg / 16), wblk, 0, stream>>>(
            rhbuf, nullptr, Uht1t, h1n, HIDg, HIDg, nullptr, zbuf, h1p, gbuf + 2 * HIDg);
        h0p = h0n;
        h1p = h1n;
    }

    // Convert all h1 states to f16 once (keeps decoder hot loop pure load+WMMA)
    cvt_f32_f16<<<dim3((SEQg * BATCHg * HIDg + tpb - 1) / tpb), tpb, 0, stream>>>(
        Hseq, Hh, SEQg * BATCHg * HIDg);

    // Decoder: logits = Hseq @ dec_W + dec_b, straight into d_out (logits-major)
    dec_gemm_wmma_f16<<<dim3(VOCABg / (16 * DEC_NT), (SEQg * BATCHg) / 16), wblk, 0, stream>>>(
        Hh, decWt, dec_b, (float*)d_out, VOCABg, HIDg);

    // final_hidden appended after logits
    float* outh = (float*)d_out + (size_t)SEQg * BATCHg * VOCABg;
    hipMemcpyAsync(outh, (const void*)h0p, (size_t)BATCHg * HIDg * 4,
                   hipMemcpyDeviceToDevice, stream);
    hipMemcpyAsync(outh + BATCHg * HIDg, Hseq + (size_t)(SEQg - 1) * BATCHg * HIDg,
                   (size_t)BATCHg * HIDg * 4, hipMemcpyDeviceToDevice, stream);
}
